// GCN_77008763617446
// MI455X (gfx1250) — matrix operand order repaired and checked by reading, not compiled, not run on hardware
//
#include <hip/hip_runtime.h>
#include <hip/hip_bf16.h>

// ---------------------------------------------------------------------------
// GCN 3-layer forward for MI455X (gfx1250, wave32).
//  - GEMMs via v_wmma_f32_16x16x32_bf16 (fp32 accumulate), LDS-staged tiles.
//  - Graph aggregation via CSR (degree -> scan -> fill), atomic-free gather:
//    one wave per node, 128 dims across 32 lanes (float4), edges serial.
//  - h buffers (25.6 MB) are L2-resident (192 MB L2); aggregation is the
//    bandwidth bottleneck (~300 MB/layer of L2 reads), GEMM flops are tiny.
// ---------------------------------------------------------------------------

typedef __attribute__((ext_vector_type(16))) __bf16 v16bf;
typedef __attribute__((ext_vector_type(8)))  float  v8f;

union Frag16 { uint4 u[2]; v16bf v; };

// ---------------------------------------------------------------------------
// GEMM: H[nrows x NOUT] = X[nrows x 128] @ W[128 x NOUT]   (bf16 in, f32 out)
// Block: 256 threads = 8 waves; each wave computes a 16-row strip.
// ---------------------------------------------------------------------------
template <int NOUT>
__global__ __launch_bounds__(256) void gemm_bf16_wmma(
    const float* __restrict__ X, const float* __restrict__ W,
    float* __restrict__ H, int nrows) {
  __shared__ __align__(16) __bf16 As[128][128];     // A tile, row-major (m,k)
  __shared__ __align__(16) __bf16 Wt[NOUT][128];    // W transposed (n,k)

  const int tid = threadIdx.x;
  const int blockRow = blockIdx.x * 128;

  // Stage X tile -> LDS as bf16 (coalesced float4 reads).
  for (int idx = tid; idx < (128 * 128) / 4; idx += 256) {
    const int r = idx >> 5;               // 32 float4 per 128-wide row
    const int c = (idx & 31) << 2;
    const int g = blockRow + r;
    float4 f = make_float4(0.f, 0.f, 0.f, 0.f);
    if (g < nrows) f = *(const float4*)(X + (size_t)g * 128 + c);
    As[r][c + 0] = (__bf16)f.x;
    As[r][c + 1] = (__bf16)f.y;
    As[r][c + 2] = (__bf16)f.z;
    As[r][c + 3] = (__bf16)f.w;
  }
  // Stage W -> LDS transposed (k-major global -> n-major LDS).
  for (int idx = tid; idx < 128 * NOUT; idx += 256) {
    const int k = idx / NOUT;
    const int n = idx % NOUT;
    Wt[n][k] = (__bf16)W[idx];
  }
  __syncthreads();

  const int wave = tid >> 5;
  const int lane = tid & 31;
  const int m0   = wave * 16;
  const bool valid = (blockRow + m0) < nrows;   // wave-uniform (N % 16 == 0)
  const int lmod  = lane & 15;
  const int ahalf = (lane < 16) ? 0 : 8;        // 16-bit A 16x32 ISA layout
  const int bhalf = (lane < 16) ? 0 : 8;        // 16-bit B 32x16 ISA layout

  if (valid) {
    // A fragments for all 4 K-steps, reused across N-tiles.
    Frag16 a[4];
    const int m = m0 + lmod;
#pragma unroll
    for (int kb = 0; kb < 4; ++kb) {
      a[kb].u[0] = *(const uint4*)&As[m][kb * 32 + ahalf];        // K = ahalf..+7
      a[kb].u[1] = *(const uint4*)&As[m][kb * 32 + ahalf + 16];   // K = 16+ahalf..+7
    }
#pragma unroll
    for (int nt = 0; nt < NOUT / 16; ++nt) {
      v8f acc = {};
      const int n = nt * 16 + lmod;
#pragma unroll
      for (int kb = 0; kb < 4; ++kb) {
        Frag16 b;
        b.u[0] = *(const uint4*)&Wt[n][kb * 32 + bhalf];          // K = bhalf..+7
        b.u[1] = *(const uint4*)&Wt[n][kb * 32 + bhalf + 16];     // K = 16+bhalf..+7
        acc = __builtin_amdgcn_wmma_f32_16x16x32_bf16(
            false, a[kb].v, false, b.v, (short)0, acc, false, false);
      }
      const int gm = blockRow + m0 + ((lane < 16) ? 0 : 8);
#pragma unroll
      for (int r = 0; r < 8; ++r) {
        H[(size_t)(gm + r) * NOUT + n] = acc[r];
      }
    }
  }
}

// ---------------------------------------------------------------------------
// CSR construction helpers
// ---------------------------------------------------------------------------
__global__ __launch_bounds__(256) void zero_int_kernel(int* __restrict__ p, int n) {
  const int i = blockIdx.x * blockDim.x + threadIdx.x;
  if (i < n) p[i] = 0;
}

__global__ __launch_bounds__(256) void degree_kernel(
    const int* __restrict__ dst, int* __restrict__ deg, int E) {
  const int e = blockIdx.x * blockDim.x + threadIdx.x;
  if (e < E) atomicAdd(&deg[dst[e]], 1);
}

__global__ __launch_bounds__(256) void dinv_kernel(
    const int* __restrict__ deg, float* __restrict__ dinv, int n) {
  const int i = blockIdx.x * blockDim.x + threadIdx.x;
  if (i < n) dinv[i] = rsqrtf((float)deg[i] + 1.0f);   // +1 self loop
}

// Single-block Hillis–Steele scan over deg -> row_ptr (exclusive start in cursor).
__global__ __launch_bounds__(1024) void scan_kernel(
    const int* __restrict__ deg, int* __restrict__ row_ptr,
    int* __restrict__ cursor, int n) {
  __shared__ int sh[1024];
  __shared__ int carry;
  const int t = threadIdx.x;
  if (t == 0) { carry = 0; row_ptr[0] = 0; }
  __syncthreads();
  for (int start = 0; start < n; start += 1024) {
    const int i = start + t;
    const int v = (i < n) ? deg[i] : 0;
    sh[t] = v;
    __syncthreads();
    for (int off = 1; off < 1024; off <<= 1) {
      const int add = (t >= off) ? sh[t - off] : 0;
      __syncthreads();
      sh[t] += add;
      __syncthreads();
    }
    const int incl = sh[t];
    const int base = carry;
    if (i < n) {
      row_ptr[i + 1] = base + incl;
      cursor[i]      = base + incl - v;   // exclusive prefix
    }
    __syncthreads();
    if (t == 1023) carry = base + incl;
    __syncthreads();
  }
}

__global__ __launch_bounds__(256) void fill_kernel(
    const int* __restrict__ src, const int* __restrict__ dst,
    const float* __restrict__ dinv, int* __restrict__ cursor,
    int* __restrict__ csr_src, float* __restrict__ csr_norm, int E) {
  const int e = blockIdx.x * blockDim.x + threadIdx.x;
  if (e < E) {
    const int s = src[e];
    const int d = dst[e];
    const int pos = atomicAdd(&cursor[d], 1);
    csr_src[pos]  = s;
    csr_norm[pos] = dinv[s] * dinv[d];
  }
}

// ---------------------------------------------------------------------------
// Atomic-free aggregation: one wave per node, dims across lanes.
// out[i] = sum_e norm[e]*H[src[e]] + dinv[i]^2 * H[i] + bias  (optional ReLU)
// ---------------------------------------------------------------------------
template <int D, bool RELU>
__global__ __launch_bounds__(256) void gather_kernel(
    const float* __restrict__ H, const int* __restrict__ row_ptr,
    const int* __restrict__ csr_src, const float* __restrict__ csr_norm,
    const float* __restrict__ dinv, const float* __restrict__ bias,
    float* __restrict__ out, int nnodes) {
  constexpr int V = D / 32;                    // floats per lane (4 or 2)
  const int wave = threadIdx.x >> 5;
  const int lane = threadIdx.x & 31;
  const int node = blockIdx.x * 8 + wave;
  if (node >= nnodes) return;

  float acc[V];
#pragma unroll
  for (int j = 0; j < V; ++j) acc[j] = 0.f;

  const int beg = row_ptr[node];
  const int end = row_ptr[node + 1];
  const int col = lane * V;

  for (int e = beg; e < end; ++e) {
    const int   s  = csr_src[e];
    const float nm = csr_norm[e];
    const float* hs = H + (size_t)s * D + col;
    if constexpr (V == 4) {
      const float4 hv = *(const float4*)hs;
      acc[0] += nm * hv.x; acc[1] += nm * hv.y;
      acc[2] += nm * hv.z; acc[3] += nm * hv.w;
    } else {
      const float2 hv = *(const float2*)hs;
      acc[0] += nm * hv.x; acc[1] += nm * hv.y;
    }
  }

  const float di = dinv[node];
  const float selfc = di * di;
  const float* hn = H + (size_t)node * D + col;
#pragma unroll
  for (int j = 0; j < V; ++j) {
    float v = acc[j] + selfc * hn[j] + bias[col + j];
    if (RELU) v = fmaxf(v, 0.f);
    out[(size_t)node * D + col + j] = v;
  }
}

// ---------------------------------------------------------------------------
// log_softmax over 64 classes: one wave per node, 2 values per lane.
// ---------------------------------------------------------------------------
__global__ __launch_bounds__(256) void log_softmax64_kernel(
    const float* __restrict__ in, float* __restrict__ out, int nnodes) {
  const int wave = threadIdx.x >> 5;
  const int lane = threadIdx.x & 31;
  const int node = blockIdx.x * 8 + wave;
  if (node >= nnodes) return;
  const float* p = in + (size_t)node * 64;
  const float a = p[lane];
  const float b = p[lane + 32];
  float m = fmaxf(a, b);
  for (int off = 16; off; off >>= 1) m = fmaxf(m, __shfl_xor(m, off, 32));
  float s = expf(a - m) + expf(b - m);
  for (int off = 16; off; off >>= 1) s += __shfl_xor(s, off, 32);
  const float ls = logf(s);
  out[(size_t)node * 64 + lane]      = a - m - ls;
  out[(size_t)node * 64 + 32 + lane] = b - m - ls;
}

// ---------------------------------------------------------------------------
// Launch
// ---------------------------------------------------------------------------
extern "C" void kernel_launch(void* const* d_in, const int* in_sizes, int n_in,
                              void* d_out, int out_size, void* d_ws, size_t ws_size,
                              hipStream_t stream) {
  const float* x  = (const float*)d_in[0];
  const float* W1 = (const float*)d_in[1];
  const float* b1 = (const float*)d_in[2];
  const float* W2 = (const float*)d_in[3];
  const float* b2 = (const float*)d_in[4];
  const float* W3 = (const float*)d_in[5];
  const float* b3 = (const float*)d_in[6];
  const int*   ei = (const int*)d_in[7];          // edge_index [2,E]

  const int N = in_sizes[0] / 128;                // 50000
  const int E = in_sizes[7] / 2;                  // 600000
  const int* src = ei;
  const int* dst = ei + E;

  // Workspace carve-up (256B-aligned slots)
  char* ws = (char*)d_ws;
  auto carve = [&](size_t bytes) -> void* {
    void* p = (void*)ws;
    ws += (bytes + 255) & ~(size_t)255;
    return p;
  };
  int*   deg      = (int*)  carve((size_t)N * 4);
  float* dinv     = (float*)carve((size_t)N * 4);
  int*   row_ptr  = (int*)  carve((size_t)(N + 1) * 4);
  int*   cursor   = (int*)  carve((size_t)N * 4);
  int*   csr_src  = (int*)  carve((size_t)E * 4);
  float* csr_norm = (float*)carve((size_t)E * 4);
  float* hA       = (float*)carve((size_t)N * 128 * 4);
  float* hB       = (float*)carve((size_t)N * 128 * 4);
  (void)ws_size; (void)n_in; (void)out_size;

  const int nThreads = 256;
  const int gN  = (N + nThreads - 1) / nThreads;
  const int gE  = (E + nThreads - 1) / nThreads;
  const int gbM = (N + 127) / 128;     // GEMM blocks (128 rows each)
  const int gbW = (N + 7) / 8;         // one wave per node kernels

  // --- CSR build (once per call; reused by all 3 layers) ---
  zero_int_kernel<<<gN, nThreads, 0, stream>>>(deg, N);
  degree_kernel<<<gE, nThreads, 0, stream>>>(dst, deg, E);
  dinv_kernel<<<gN, nThreads, 0, stream>>>(deg, dinv, N);
  scan_kernel<<<1, 1024, 0, stream>>>(deg, row_ptr, cursor, N);
  fill_kernel<<<gE, nThreads, 0, stream>>>(src, dst, dinv, cursor,
                                           csr_src, csr_norm, E);

  // --- Layer 1: x(128) -> 128, ReLU ---
  gemm_bf16_wmma<128><<<gbM, nThreads, 0, stream>>>(x, W1, hA, N);
  gather_kernel<128, true><<<gbW, nThreads, 0, stream>>>(
      hA, row_ptr, csr_src, csr_norm, dinv, b1, hB, N);

  // --- Layer 2: 128 -> 128, ReLU ---
  gemm_bf16_wmma<128><<<gbM, nThreads, 0, stream>>>(hB, W2, hA, N);
  gather_kernel<128, true><<<gbW, nThreads, 0, stream>>>(
      hA, row_ptr, csr_src, csr_norm, dinv, b2, hB, N);

  // --- Layer 3: 128 -> 64, no ReLU, then log_softmax ---
  gemm_bf16_wmma<64><<<gbM, nThreads, 0, stream>>>(hB, W3, hA, N);
  gather_kernel<64, false><<<gbW, nThreads, 0, stream>>>(
      hA, row_ptr, csr_src, csr_norm, dinv, b3, hB, N);
  log_softmax64_kernel<<<gbW, nThreads, 0, stream>>>(hB, (float*)d_out, N);
}
